// SrnmSpmm_49443663512204
// MI455X (gfx1250) — matrix operand order repaired and checked by reading, not compiled, not run on hardware
//
#include <hip/hip_runtime.h>

typedef __attribute__((ext_vector_type(16))) __bf16 v16bf;
typedef __attribute__((ext_vector_type(8)))  __bf16 v8bf;
typedef __attribute__((ext_vector_type(2)))  __bf16 v2bf;
typedef __attribute__((ext_vector_type(8)))  float  v8f;
typedef __attribute__((ext_vector_type(4)))  float  v4f;

#define BM 128
#define BN 128
#define GPS 16               // groups of 8 source cols per K-step -> BK = 32 bf16 K values
#define LDS_STRIDE 40        // bf16 per LDS row (80B = 20 dwords; bank-conflict free)

// out[m, n] = sum_{k<D/4} x[m, d(k)] * W[n, d(k)] + bias[n]
// d(k) = (k>>1)*8 + (k&1)*2   (fixed 2:8 keep pattern [1,0,1,0,0,0,0,0])
__global__ __launch_bounds__(256)
void srnm_spmm_wmma(const float* __restrict__ x,
                    const float* __restrict__ W,
                    const float* __restrict__ bias,
                    float* __restrict__ out,
                    int M, int N, int D)
{
    __shared__ __bf16 sA[2][BM * LDS_STRIDE];   // 2 x 10 KB
    __shared__ __bf16 sB[2][BN * LDS_STRIDE];   // 2 x 10 KB

    const int tid   = threadIdx.x;
    const int lane  = tid & 31;
    const int wave  = tid >> 5;     // 0..7
    const int waveM = wave >> 2;    // 0..1 -> 64 output rows
    const int waveN = wave & 3;     // 0..3 -> 32 output cols

    const int mBase = blockIdx.y * BM;
    const int nBase = blockIdx.x * BN;

    // loader mapping: 256 threads x 8 iters cover 128 rows x 16 groups
    const int ldGroup = tid & (GPS - 1);   // 0..15
    const int ldRow0  = tid >> 4;          // 0..15

    const float* xRow = x + (size_t)(mBase + ldRow0) * D + (size_t)ldGroup * 8;
    const float* wRow = W + (size_t)(nBase + ldRow0) * D + (size_t)ldGroup * 8;

    const v8f vzero = {0.f, 0.f, 0.f, 0.f, 0.f, 0.f, 0.f, 0.f};
    v8f acc[4][2];
    #pragma unroll
    for (int mi = 0; mi < 4; ++mi)
        #pragma unroll
        for (int ni = 0; ni < 2; ++ni)
            acc[mi][ni] = vzero;

    const int ksteps = D / (8 * GPS);      // 4096/128 = 32

    // in-flight prefetch registers (raw f32 quads; convert after loads land)
    v4f qa[8], qb[8];

    auto prefetch = [&](int kt) {
        const size_t koff = (size_t)kt * GPS * 8;
        #pragma unroll
        for (int i = 0; i < 8; ++i) {
            qa[i] = *(const v4f*)(xRow + koff + (size_t)i * 16 * D);
            qb[i] = *(const v4f*)(wRow + koff + (size_t)i * 16 * D);
        }
    };
    auto stage = [&](int buf) {
        #pragma unroll
        for (int i = 0; i < 8; ++i) {
            v2bf pa = { (__bf16)qa[i][0], (__bf16)qa[i][2] };  // keep {0,2} of group
            v2bf pb = { (__bf16)qb[i][0], (__bf16)qb[i][2] };
            *(v2bf*)&sA[buf][(ldRow0 + i * 16) * LDS_STRIDE + 2 * ldGroup] = pa;
            *(v2bf*)&sB[buf][(ldRow0 + i * 16) * LDS_STRIDE + 2 * ldGroup] = pb;
        }
    };

    // fragment addressing (CDNA5 16-bit WMMA layouts)
    const int mr  = lane & 15;
    const int kcA = (lane >> 4) * 8;    // A: two 8-elem chunks at +0 / +16
    const int kcB = (lane >> 4) * 8;    // B: two 8-elem chunks at +0 / +16

    // ---- software pipeline: prefetch(kt+1) overlaps WMMAs of kt ----
    prefetch(0);
    stage(0);
    __syncthreads();

    for (int kt = 0; kt < ksteps; ++kt) {
        const int  buf  = kt & 1;
        const bool more = (kt + 1) < ksteps;

        if (more) prefetch(kt + 1);   // global loads in flight during compute

        v16bf afrag[4], bfrag[2];
        #pragma unroll
        for (int mi = 0; mi < 4; ++mi) {
            const __bf16* p = &sA[buf][(waveM * 64 + mi * 16 + mr) * LDS_STRIDE + kcA];
            v8bf lo = *(const v8bf*)p;
            v8bf hi = *(const v8bf*)(p + 16);
            afrag[mi] = __builtin_shufflevector(lo, hi,
                0,1,2,3,4,5,6,7,8,9,10,11,12,13,14,15);
        }
        #pragma unroll
        for (int ni = 0; ni < 2; ++ni) {
            const __bf16* p = &sB[buf][(waveN * 32 + ni * 16 + mr) * LDS_STRIDE + kcB];
            v8bf lo = *(const v8bf*)p;
            v8bf hi = *(const v8bf*)(p + 16);
            bfrag[ni] = __builtin_shufflevector(lo, hi,
                0,1,2,3,4,5,6,7,8,9,10,11,12,13,14,15);
        }

        #pragma unroll
        for (int mi = 0; mi < 4; ++mi)
            #pragma unroll
            for (int ni = 0; ni < 2; ++ni)
                acc[mi][ni] = __builtin_amdgcn_wmma_f32_16x16x32_bf16(
                    false, afrag[mi], false, bfrag[ni],
                    (short)0, acc[mi][ni], false, false);

        if (more) stage(buf ^ 1);     // write NEXT buffer; current reads already done
        __syncthreads();              // single barrier per iteration
    }

    // ---- epilogue: C/D layout lane = N%16, VGPR r -> M = r + 8*(lane>=16) ----
    const int colLo = nBase + waveN * 32 + (lane & 15);
    const int rsel  = (lane >> 4) * 8;
    #pragma unroll
    for (int ni = 0; ni < 2; ++ni) {
        const int c  = colLo + ni * 16;
        const float bv = bias[c];
        #pragma unroll
        for (int mi = 0; mi < 4; ++mi) {
            #pragma unroll
            for (int r = 0; r < 8; ++r) {
                const int rowg = mBase + waveM * 64 + mi * 16 + rsel + r;
                out[(size_t)rowg * N + c] = acc[mi][ni][r] + bv;
            }
        }
    }
}

extern "C" void kernel_launch(void* const* d_in, const int* in_sizes, int n_in,
                              void* d_out, int out_size, void* d_ws, size_t ws_size,
                              hipStream_t stream) {
    (void)n_in; (void)out_size; (void)d_ws; (void)ws_size;
    const float* x    = (const float*)d_in[0];
    const float* W    = (const float*)d_in[1];
    const float* bias = (const float*)d_in[2];
    float* out        = (float*)d_out;

    const int D = in_sizes[2];          // 4096 (feature dim)
    const int N = in_sizes[1] / D;      // 4096
    const int M = in_sizes[0] / D;      // 8192 = B*S

    dim3 grid(N / BN, M / BM);          // (32, 64)
    srnm_spmm_wmma<<<grid, 256, 0, stream>>>(x, W, bias, out, M, N, D);
}
